// GRUSeq2Seq_68994354643251
// MI455X (gfx1250) — matrix operand order repaired and checked by reading, not compiled, not run on hardware
//
#include <hip/hip_runtime.h>

typedef __bf16 bf16;
typedef __attribute__((ext_vector_type(16))) bf16 v16bf;
typedef __attribute__((ext_vector_type(8)))  bf16 v8bf;
typedef __attribute__((ext_vector_type(8)))  float v8f;
typedef __attribute__((ext_vector_type(4)))  int v4i;

#define LANE (threadIdx.x & 31)

#if __has_builtin(__builtin_amdgcn_global_load_async_to_lds_b128)
#define HAS_ASYNC_LDS 1
#else
#define HAS_ASYNC_LDS 0
#endif

#if HAS_ASYNC_LDS
typedef __attribute__((address_space(1))) v4i gas_v4i;
typedef __attribute__((address_space(3))) v4i las_v4i;
__device__ __forceinline__ void async_copy16(const bf16* g, bf16* l) {
  v4i* gp = (v4i*)g;   // strips const; generic pointer
  v4i* lp = (v4i*)l;
  __builtin_amdgcn_global_load_async_to_lds_b128((gas_v4i*)gp, (las_v4i*)lp, 0, 0);
}
#if __has_builtin(__builtin_amdgcn_s_wait_asynccnt)
#define WAIT_ASYNC(n) __builtin_amdgcn_s_wait_asynccnt(n)
#else
#define WAIT_ASYNC(n) asm volatile("s_wait_asynccnt %0" ::"n"(n) : "memory")
#endif
#endif

__device__ __forceinline__ v8f vzero() {
  v8f z = {0.f,0.f,0.f,0.f,0.f,0.f,0.f,0.f};
  return z;
}

__device__ __forceinline__ v16bf pack16(v8bf lo, v8bf hi) {
  return __builtin_shufflevector(lo, hi, 0,1,2,3,4,5,6,7,8,9,10,11,12,13,14,15);
}

// A-tile 16x32 (MxK), row-major source, ISA A layout: lane<16 -> K [0..7],[16..23]
__device__ __forceinline__ v16bf load_a_bf16(const bf16* s, int ld) {
  int r  = LANE & 15;
  int kh = (LANE >> 4) << 3;          // 0 or 8
  const bf16* p = s + r * ld + kh;
  v8bf lo = *(const v8bf*)p;          // K kh..kh+7
  v8bf hi = *(const v8bf*)(p + 16);   // K kh+16..kh+23
  return pack16(lo, hi);
}

// A-tile from fp32 source (LDS h state) with on-the-fly bf16 convert
__device__ __forceinline__ v16bf load_a_f32(const float* s, int ld) {
  int r  = LANE & 15;
  int kh = (LANE >> 4) << 3;
  const float* p = s + r * ld + kh;
  v16bf t;
#pragma unroll
  for (int i = 0; i < 8; ++i) { t[i] = (bf16)p[i]; t[i + 8] = (bf16)p[i + 16]; }
  return t;
}

// B-tile 32x16 (KxN) from row-major (N,K) weight; ISA B layout: lane<16 -> K [0..7],[16..23]
__device__ __forceinline__ v16bf load_b_bf16(const bf16* s, int ld) {
  int n  = LANE & 15;
  int k0 = (LANE >> 4) << 3;          // 0 or 8
  const bf16* p = s + n * ld + k0;
  v8bf lo = *(const v8bf*)p;
  v8bf hi = *(const v8bf*)(p + 16);
  return pack16(lo, hi);
}

__device__ __forceinline__ v8f wmma_bf16(v16bf a, v16bf b, v8f c) {
  return __builtin_amdgcn_wmma_f32_16x16x32_bf16(false, a, false, b, (short)0, c,
                                                 false, false);
}

__device__ __forceinline__ float sigm(float x) { return 1.0f / (1.0f + __expf(-x)); }

// ---------------------------------------------------------------- utilities
__global__ void cvt_f32_bf16(const float* __restrict__ in, bf16* __restrict__ out, int n) {
  int i = blockIdx.x * 256 + threadIdx.x;
  if (i < n) out[i] = (bf16)in[i];
}

// Split W_ih_d (1536 x 768) into embedding part (1536x256) and context part (1536x512)
__global__ void split_wihd(const float* __restrict__ W, bf16* __restrict__ Wemb,
                           bf16* __restrict__ Wctx) {
  int idx = blockIdx.x * 256 + threadIdx.x;
  if (idx >= 1536 * 768) return;
  int row = idx / 768, col = idx % 768;
  bf16 v = (bf16)W[idx];
  if (col < 256) Wemb[row * 256 + col] = v;
  else           Wctx[row * 512 + (col - 256)] = v;
}

// x_enc[(s*64+b)*256 + e] = emb_enc[src[b][s]][e]  (bf16)
__global__ void embed_enc(const int* __restrict__ src, const float* __restrict__ emb,
                          bf16* __restrict__ out) {
  int sb = blockIdx.x;                 // s*64 + b
  int s = sb >> 6, b = sb & 63;
  int tok = src[b * 32 + s];
  out[(size_t)sb * 256 + threadIdx.x] = (bf16)emb[(size_t)tok * 256 + threadIdx.x];
}

// x_dec[(t*64+b)*256 + e] = emb_dec[trg[b][t]][e], t in [0,31)
__global__ void embed_dec(const int* __restrict__ trg, const float* __restrict__ emb,
                          bf16* __restrict__ out) {
  int tb = blockIdx.x;                 // t*64 + b
  int t = tb >> 6, b = tb & 63;
  int tok = trg[b * 32 + t];
  out[(size_t)tb * 256 + threadIdx.x] = (bf16)emb[(size_t)tok * 256 + threadIdx.x];
}

// out[b][0][:] = 0
__global__ void zero_t0(float* __restrict__ out) {
  int i = blockIdx.x * 256 + threadIdx.x;
  if (i >= 64 * 32000) return;
  int b = i / 32000, v = i % 32000;
  out[(size_t)b * 32 * 32000 + v] = 0.f;
}

// ---------------------------------------------------------------- generic WMMA GEMM
// C(M,N) = A(M,K)bf16 * B(N,K)^T bf16 (+bias). mode 0: row-major C.
// mode 1: row = step*64+b -> d_out[b][step+1][col]  (B=64,T=32,V=32000)
// K-chunks of 64 staged to LDS with double-buffered GLOBAL_LOAD_ASYNC_TO_LDS_B128
// (ASYNCcnt-tracked), overlapping next-chunk fetch with current-chunk WMMA.
__global__ void __launch_bounds__(256)
gemm_bf16(const bf16* __restrict__ A, const bf16* __restrict__ Bw,
          const float* __restrict__ bias, float* __restrict__ C,
          int M, int N, int K, int mode) {
  __shared__ __align__(16) bf16 sA[2][64 * 64];
  __shared__ __align__(16) bf16 sB[2][128 * 64];
  int tid = threadIdx.x, wave = tid >> 5;
  int m0 = blockIdx.y * 64, n0 = blockIdx.x * 128;

  v8f acc[4];
#pragma unroll
  for (int t = 0; t < 4; ++t) acc[t] = vzero();

  int ea = tid * 16, ar = ea >> 6, ac = ea & 63;   // A: 16 bf16 (2 x b128) / thread
  int eb = tid * 32, br = eb >> 6, bc = eb & 63;   // B: 32 bf16 (4 x b128) / thread
  const bf16* gA = A  + (size_t)(m0 + ar) * K + ac;
  const bf16* gB = Bw + (size_t)(n0 + br) * K + bc;
  int nchunk = K >> 6;

#if HAS_ASYNC_LDS
  auto issue = [&](int buf, int kc) {
    async_copy16(gA + kc,      sA[buf] + ea);
    async_copy16(gA + kc + 8,  sA[buf] + ea + 8);
    async_copy16(gB + kc,      sB[buf] + eb);
    async_copy16(gB + kc + 8,  sB[buf] + eb + 8);
    async_copy16(gB + kc + 16, sB[buf] + eb + 16);
    async_copy16(gB + kc + 24, sB[buf] + eb + 24);
  };
  issue(0, 0);
  for (int c = 0; c < nchunk; ++c) {
    int buf = c & 1;
    if (c + 1 < nchunk) {           // prefetch next chunk into other buffer
      issue(buf ^ 1, (c + 1) << 6);
      WAIT_ASYNC(6);                // in-order: current chunk's 6 copies done
    } else {
      WAIT_ASYNC(0);
    }
    __syncthreads();
#pragma unroll
    for (int k = 0; k < 2; ++k) {
      v16bf bt = load_b_bf16(sB[buf] + wave * 16 * 64 + k * 32, 64);
#pragma unroll
      for (int tm = 0; tm < 4; ++tm) {
        v16bf at = load_a_bf16(sA[buf] + tm * 16 * 64 + k * 32, 64);
        acc[tm] = wmma_bf16(at, bt, acc[tm]);
      }
    }
    __syncthreads();
  }
#else
  for (int c = 0; c < nchunk; ++c) {
    int kc = c << 6;
    const uint4* ga = (const uint4*)(gA + kc);
    *(uint4*)(sA[0] + ea)     = ga[0];
    *(uint4*)(sA[0] + ea + 8) = ga[1];
    const uint4* gb = (const uint4*)(gB + kc);
    *(uint4*)(sB[0] + eb)      = gb[0];
    *(uint4*)(sB[0] + eb + 8)  = gb[1];
    *(uint4*)(sB[0] + eb + 16) = gb[2];
    *(uint4*)(sB[0] + eb + 24) = gb[3];
    __syncthreads();
#pragma unroll
    for (int k = 0; k < 2; ++k) {
      v16bf bt = load_b_bf16(sB[0] + wave * 16 * 64 + k * 32, 64);
#pragma unroll
      for (int tm = 0; tm < 4; ++tm) {
        v16bf at = load_a_bf16(sA[0] + tm * 16 * 64 + k * 32, 64);
        acc[tm] = wmma_bf16(at, bt, acc[tm]);
      }
    }
    __syncthreads();
  }
#endif

  int crow0 = (LANE >> 4) << 3, ccol = LANE & 15;
  int col = n0 + wave * 16 + ccol;
  float bv = bias ? bias[col] : 0.f;
#pragma unroll
  for (int tm = 0; tm < 4; ++tm) {
#pragma unroll
    for (int i = 0; i < 8; ++i) {
      int row = m0 + tm * 16 + crow0 + i;
      float v = acc[tm][i] + bv;
      if (mode == 0) {
        C[(size_t)row * N + col] = v;
      } else {
        int b = row & 63, st = row >> 6;
        C[(size_t)b * (32 * 32000) + (size_t)(st + 1) * 32000 + col] = v;
      }
    }
  }
}

// ---------------------------------------------------------------- encoder (persistent)
// One workgroup per direction. h (64x256 fp32) double-buffered in LDS.
// gh = h @ W_hh^T via WMMA; full gate math in registers (one column per lane).
__global__ void __launch_bounds__(512)
gru_encoder(const float* __restrict__ gi_fwd, const float* __restrict__ gi_bwd,
            const bf16* __restrict__ Whh_f, const bf16* __restrict__ Whh_b,
            const float* __restrict__ bhh_f, const float* __restrict__ bhh_b,
            float* __restrict__ context) {
  extern __shared__ float smem[];
  const int HSZ = 64 * 256;
  int dir = blockIdx.x;
  const float* gi  = dir ? gi_bwd : gi_fwd;
  const bf16*  Whh = dir ? Whh_b : Whh_f;
  const float* bhh = dir ? bhh_b : bhh_f;
  int tid = threadIdx.x, wave = tid >> 5;
  for (int i = tid; i < 2 * HSZ; i += 512) smem[i] = 0.f;
  __syncthreads();

  int cur = 0;
  int crow0 = (LANE >> 4) << 3, ccol = LANE & 15;
  for (int step = 0; step < 32; ++step) {
    const float* hc = smem + (cur ? HSZ : 0);
    float*       hn = smem + (cur ? 0 : HSZ);
    int sg = dir ? (31 - step) : step;
    const float* gis = gi + (size_t)sg * 64 * 768;
#pragma unroll
    for (int q = 0; q < 4; ++q) {
      int u = wave * 4 + q;
      int m = u >> 4, j = u & 15;        // m: 4 M-tiles, j: 16 N-tiles per gate
      v8f ar = vzero(), az = vzero(), an = vzero();
#pragma unroll
      for (int k = 0; k < 8; ++k) {
        v16bf a  = load_a_f32(hc + m * 16 * 256 + k * 32, 256);
        v16bf b0 = load_b_bf16(Whh + (size_t)(        j * 16) * 256 + k * 32, 256);
        v16bf b1 = load_b_bf16(Whh + (size_t)( 256 + j * 16) * 256 + k * 32, 256);
        v16bf b2 = load_b_bf16(Whh + (size_t)( 512 + j * 16) * 256 + k * 32, 256);
        ar = wmma_bf16(a, b0, ar);
        az = wmma_bf16(a, b1, az);
        an = wmma_bf16(a, b2, an);
      }
      int col = j * 16 + ccol;
      float br = bhh[col], bz = bhh[256 + col], bn = bhh[512 + col];
#pragma unroll
      for (int i = 0; i < 8; ++i) {
        int row = m * 16 + crow0 + i;
        const float* g = gis + (size_t)row * 768;
        float r = sigm(g[col]        + ar[i] + br);
        float z = sigm(g[256 + col]  + az[i] + bz);
        float n = tanhf(g[512 + col] + r * (an[i] + bn));
        float ho = hc[row * 256 + col];
        hn[row * 256 + col] = (1.f - z) * n + z * ho;
      }
    }
    __syncthreads();
    cur ^= 1;
  }
  const float* hf = smem + (cur ? HSZ : 0);
  for (int i = tid; i < HSZ; i += 512) {
    int b = i >> 8, c = i & 255;
    context[(size_t)b * 512 + dir * 256 + c] = hf[i];
  }
}

// ---------------------------------------------------------------- decoder (persistent)
// One workgroup. h (64x512 fp32) double-buffered in LDS (256 KB <= 320 KB).
// gi_total = gi_emb[t] + gi_ctx (context contribution precomputed once).
// Writes h_t in bf16 to h_all for the batched FC GEMM.
__global__ void __launch_bounds__(512)
gru_decoder(const float* __restrict__ gi_emb, const float* __restrict__ gi_ctx,
            const bf16* __restrict__ Whh, const float* __restrict__ bhh,
            const float* __restrict__ context, bf16* __restrict__ h_all) {
  extern __shared__ float smem[];
  const int HSZ = 64 * 512;
  int tid = threadIdx.x, wave = tid >> 5;
  for (int i = tid; i < HSZ; i += 512) smem[i] = context[i];
  __syncthreads();

  int cur = 0;
  int crow0 = (LANE >> 4) << 3, ccol = LANE & 15;
  for (int t = 0; t < 31; ++t) {
    const float* hc = smem + (cur ? HSZ : 0);
    float*       hn = smem + (cur ? 0 : HSZ);
    const float* gis = gi_emb + (size_t)t * 64 * 1536;
#pragma unroll
    for (int q = 0; q < 8; ++q) {
      int u = wave * 8 + q;
      int m = u >> 5, j = u & 31;        // m: 4 M-tiles, j: 32 N-tiles per gate
      v8f ar = vzero(), az = vzero(), an = vzero();
#pragma unroll 4
      for (int k = 0; k < 16; ++k) {
        v16bf a  = load_a_f32(hc + m * 16 * 512 + k * 32, 512);
        v16bf b0 = load_b_bf16(Whh + (size_t)(         j * 16) * 512 + k * 32, 512);
        v16bf b1 = load_b_bf16(Whh + (size_t)( 512  + j * 16) * 512 + k * 32, 512);
        v16bf b2 = load_b_bf16(Whh + (size_t)( 1024 + j * 16) * 512 + k * 32, 512);
        ar = wmma_bf16(a, b0, ar);
        az = wmma_bf16(a, b1, az);
        an = wmma_bf16(a, b2, an);
      }
      int col = j * 16 + ccol;
      float br = bhh[col], bz = bhh[512 + col], bn = bhh[1024 + col];
#pragma unroll
      for (int i = 0; i < 8; ++i) {
        int row = m * 16 + crow0 + i;
        const float* ge = gis + (size_t)row * 1536;
        const float* gc = gi_ctx + (size_t)row * 1536;
        float r = sigm(ge[col]         + gc[col]         + ar[i] + br);
        float z = sigm(ge[512 + col]   + gc[512 + col]   + az[i] + bz);
        float n = tanhf(ge[1024 + col] + gc[1024 + col]  + r * (an[i] + bn));
        float ho = hc[row * 512 + col];
        float hv = (1.f - z) * n + z * ho;
        hn[row * 512 + col] = hv;
        h_all[((size_t)t * 64 + row) * 512 + col] = (bf16)hv;
      }
    }
    __syncthreads();
    cur ^= 1;
  }
}

// ---------------------------------------------------------------- host
extern "C" void kernel_launch(void* const* d_in, const int* in_sizes, int n_in,
                              void* d_out, int out_size, void* d_ws, size_t ws_size,
                              hipStream_t stream) {
  const int*   src     = (const int*)d_in[0];
  const int*   trg     = (const int*)d_in[1];
  const float* emb_enc = (const float*)d_in[2];
  const float* W_ih_f  = (const float*)d_in[3];
  const float* W_hh_f  = (const float*)d_in[4];
  const float* b_ih_f  = (const float*)d_in[5];
  const float* b_hh_f  = (const float*)d_in[6];
  const float* W_ih_b  = (const float*)d_in[7];
  const float* W_hh_b  = (const float*)d_in[8];
  const float* b_ih_b  = (const float*)d_in[9];
  const float* b_hh_b  = (const float*)d_in[10];
  const float* emb_dec = (const float*)d_in[11];
  const float* W_ih_d  = (const float*)d_in[12];
  const float* W_hh_d  = (const float*)d_in[13];
  const float* b_ih_d  = (const float*)d_in[14];
  const float* b_hh_d  = (const float*)d_in[15];
  const float* fc_W    = (const float*)d_in[16];
  const float* fc_b    = (const float*)d_in[17];
  float* out = (float*)d_out;

  char* ws = (char*)d_ws;
  size_t off = 0;
  auto alloc = [&](size_t bytes) -> void* {
    void* p = ws + off;
    off = (off + bytes + 255) & ~(size_t)255;
    return p;
  };

  bf16* Wihf_bf   = (bf16*)alloc(768 * 256 * 2);
  bf16* Whhf_bf   = (bf16*)alloc(768 * 256 * 2);
  bf16* Wihb_bf   = (bf16*)alloc(768 * 256 * 2);
  bf16* Whhb_bf   = (bf16*)alloc(768 * 256 * 2);
  bf16* Wihd_e_bf = (bf16*)alloc((size_t)1536 * 256 * 2);
  bf16* Wihd_c_bf = (bf16*)alloc((size_t)1536 * 512 * 2);
  bf16* Whhd_bf   = (bf16*)alloc((size_t)1536 * 512 * 2);
  bf16* fcW_bf    = (bf16*)alloc((size_t)32000 * 512 * 2);
  bf16* x_enc     = (bf16*)alloc((size_t)2048 * 256 * 2);
  float* gi_f     = (float*)alloc((size_t)2048 * 768 * 4);
  float* gi_b     = (float*)alloc((size_t)2048 * 768 * 4);
  float* context  = (float*)alloc((size_t)64 * 512 * 4);
  bf16* ctx_bf    = (bf16*)alloc((size_t)64 * 512 * 2);
  bf16* x_dec     = (bf16*)alloc((size_t)1984 * 256 * 2);
  float* gi_emb   = (float*)alloc((size_t)1984 * 1536 * 4);
  float* gi_ctx   = (float*)alloc((size_t)64 * 1536 * 4);
  bf16* h_all     = (bf16*)alloc((size_t)1984 * 512 * 2);

  auto cvt = [&](const float* in, bf16* o, int n) {
    cvt_f32_bf16<<<(n + 255) / 256, 256, 0, stream>>>(in, o, n);
  };

  // weight conversions
  cvt(W_ih_f, Wihf_bf, 768 * 256);
  cvt(W_hh_f, Whhf_bf, 768 * 256);
  cvt(W_ih_b, Wihb_bf, 768 * 256);
  cvt(W_hh_b, Whhb_bf, 768 * 256);
  cvt(W_hh_d, Whhd_bf, 1536 * 512);
  cvt(fc_W, fcW_bf, 32000 * 512);
  split_wihd<<<(1536 * 768 + 255) / 256, 256, 0, stream>>>(W_ih_d, Wihd_e_bf, Wihd_c_bf);

  // encoder input path: gather + batched gi GEMMs (all timesteps at once)
  embed_enc<<<2048, 256, 0, stream>>>(src, emb_enc, x_enc);
  gemm_bf16<<<dim3(768 / 128, 2048 / 64), 256, 0, stream>>>(
      x_enc, Wihf_bf, b_ih_f, gi_f, 2048, 768, 256, 0);
  gemm_bf16<<<dim3(768 / 128, 2048 / 64), 256, 0, stream>>>(
      x_enc, Wihb_bf, b_ih_b, gi_b, 2048, 768, 256, 0);

  // sequential bidirectional encoder (1 workgroup per direction)
  gru_encoder<<<2, 512, 2 * 64 * 256 * sizeof(float), stream>>>(
      gi_f, gi_b, Whhf_bf, Whhb_bf, b_hh_f, b_hh_b, context);

  // decoder input path
  cvt(context, ctx_bf, 64 * 512);
  embed_dec<<<1984, 256, 0, stream>>>(trg, emb_dec, x_dec);
  gemm_bf16<<<dim3(1536 / 128, 1984 / 64), 256, 0, stream>>>(
      x_dec, Wihd_e_bf, b_ih_d, gi_emb, 1984, 1536, 256, 0);
  gemm_bf16<<<dim3(1536 / 128, 64 / 64), 256, 0, stream>>>(
      ctx_bf, Wihd_c_bf, nullptr, gi_ctx, 64, 1536, 512, 0);

  // sequential decoder (1 workgroup, 256 KB LDS double-buffered h)
  gru_decoder<<<1, 512, 2 * 64 * 512 * sizeof(float), stream>>>(
      gi_emb, gi_ctx, Whhd_bf, b_hh_d, context, h_all);

  // outputs: t=0 zeros, then one batched FC GEMM scattered into (B,T,V)
  zero_t0<<<(64 * 32000 + 255) / 256, 256, 0, stream>>>(out);
  gemm_bf16<<<dim3(32000 / 128, 1984 / 64), 256, 0, stream>>>(
      h_all, fcW_bf, fc_b, out, 1984, 32000, 512, 1);
}
